// DIEPGraphConv_10677288698373
// MI455X (gfx1250) — compile-verified, matrix operand order checked by reading, not run on hardware
//
#include <hip/hip_runtime.h>
#include <hip/hip_bf16.h>

typedef __attribute__((ext_vector_type(16))) __bf16 v16bf;
typedef __attribute__((ext_vector_type(8)))  __bf16 v8bf;
typedef __attribute__((ext_vector_type(8)))  float  v8f;

#define GDIM   64
#define GD_IN  192
#define WAVES  8

// ---- LDS layout (bytes) ----
#define SZ_W0   (GD_IN * GDIM * 2)        // 24576  (192x64 bf16 fragments)
#define SZ_W1   (GDIM * GDIM * 2)         // 8192   (64x64 bf16 fragments)
#define OFF_L0  0
#define OFF_G0  (OFF_L0 + SZ_W0)
#define OFF_L1  (OFF_G0 + SZ_W0)
#define OFF_G1  (OFF_L1 + SZ_W1)
#define OFF_WW  (OFF_G1 + SZ_W1)          // 9*64 f32 = 2304
#define OFF_B   (OFF_WW + 9 * GDIM * 4)   // 4*64 f32 = 1024
#define OFF_WV  (OFF_B + 4 * GDIM * 4)    // = 68864, 16B aligned
#define X_SZ    (16 * GD_IN * 2)          // 6144  x tile bf16
#define W_SZ    (16 * GDIM * 4)           // 4096  rbf-weight tile f32
#define H_SZ    (16 * GDIM * 2)           // 2048  hidden tile bf16
#define WV_SZ   (X_SZ + W_SZ + H_SZ)      // 12288 per wave
#define LDS_TOTAL (OFF_WV + WAVES * WV_SZ)  // 167168 B < 320KB/WGP

__device__ __forceinline__ float silu_f(float x) { return x / (1.f + __expf(-x)); }
__device__ __forceinline__ float sigm_f(float x) { return 1.f / (1.f + __expf(-x)); }

__device__ __forceinline__ v16bf cat8(v8bf lo, v8bf hi) {
  return __builtin_shufflevector(lo, hi, 0,1,2,3,4,5,6,7,8,9,10,11,12,13,14,15);
}

// A-fragment (16-bit, 16x32): lane holds row m=lane&15; element i -> k =
// 16*(i>>3) + 8*(lane>>4) + (i&7). Two contiguous 16B runs per lane.
__device__ __forceinline__ v16bf load_afrag(const __bf16* row, int kbase, int half) {
  v8bf lo = *(const v8bf*)(row + kbase + half * 8);
  v8bf hi = *(const v8bf*)(row + kbase + 16 + half * 8);
  return cat8(lo, hi);
}

// B fragments are pre-swizzled in LDS: contiguous 16 bf16 per lane per frag.
__device__ __forceinline__ v16bf load_bfrag(const __bf16* frag, int kt, int nt, int lane) {
  const __bf16* p = frag + (((kt * 4 + nt) * 32) + lane) * 16;
  v8bf lo = *(const v8bf*)p;
  v8bf hi = *(const v8bf*)(p + 8);
  return cat8(lo, hi);
}

// Convert fp32 row-major weight [kdim x 64] into WMMA B-fragment layout (bf16).
__device__ void stage_frag(const float* __restrict__ W, __bf16* dst,
                           int kdim, int tid, int nthr) {
  for (int idx = tid; idx < kdim * GDIM; idx += nthr) {
    int k  = idx / GDIM;
    int n  = idx - k * GDIM;
    int kt = k >> 5, k32 = k & 31;
    int half = (k32 >> 3) & 1;
    int i  = ((k32 >> 4) << 3) | (k32 & 7);
    int ln = (n & 15) | (half << 4);
    int nt = n >> 4;
    dst[(((kt * 4 + nt) * 32) + ln) * 16 + i] = (__bf16)W[idx];
  }
}

// One branch of the GatedMLP on a 16-row tile.
// res[nt][r] <-> row m = r + 8*(lane>>4), col n = nt*16 + (lane&15)  (C layout)
__device__ __forceinline__ void mlp_branch(
    const __bf16* xlds, __bf16* hlds,
    const __bf16* w0f, const float* b0,
    const __bf16* w1f, const float* b1,
    bool sig2, int lane, v8f res[4]) {
  const int half = lane >> 4;
  const int ln   = lane & 15;
  const __bf16* xrow = xlds + ln * GD_IN;

  v8f acc[4] = {};
  #pragma unroll
  for (int kt = 0; kt < 6; ++kt) {            // K = 192
    v16bf a = load_afrag(xrow, kt * 32, half);
    #pragma unroll
    for (int nt = 0; nt < 4; ++nt) {          // N = 64
      v16bf b = load_bfrag(w0f, kt, nt, lane);
      acc[nt] = __builtin_amdgcn_wmma_f32_16x16x32_bf16(
          false, a, false, b, (short)0, acc[nt], false, false);
    }
  }
  // bias + SiLU -> hidden tile (row-major bf16 in LDS)
  #pragma unroll
  for (int nt = 0; nt < 4; ++nt) {
    float bb = b0[nt * 16 + ln];
    #pragma unroll
    for (int r = 0; r < 8; ++r) {
      float v = silu_f(acc[nt][r] + bb);
      hlds[(r + half * 8) * GDIM + nt * 16 + ln] = (__bf16)v;
    }
  }
  asm volatile("s_wait_dscnt 0" ::: "memory");

  v8f acc2[4] = {};
  const __bf16* hrow = hlds + ln * GDIM;
  #pragma unroll
  for (int kt = 0; kt < 2; ++kt) {            // K = 64
    v16bf a = load_afrag(hrow, kt * 32, half);
    #pragma unroll
    for (int nt = 0; nt < 4; ++nt) {
      v16bf b = load_bfrag(w1f, kt, nt, lane);
      acc2[nt] = __builtin_amdgcn_wmma_f32_16x16x32_bf16(
          false, a, false, b, (short)0, acc2[nt], false, false);
    }
  }
  #pragma unroll
  for (int nt = 0; nt < 4; ++nt) {
    float bb = b1[nt * 16 + ln];
    #pragma unroll
    for (int r = 0; r < 8; ++r) {
      float v = acc2[nt][r] + bb;
      res[nt][r] = sig2 ? sigm_f(v) : silu_f(v);
    }
  }
}

template<bool ATOMIC>
__global__ __launch_bounds__(WAVES * 32)
void gmlp_kernel(const float* __restrict__ w0l, const float* __restrict__ b0l,
                 const float* __restrict__ w1l, const float* __restrict__ b1l,
                 const float* __restrict__ w0g, const float* __restrict__ b0g,
                 const float* __restrict__ w1g, const float* __restrict__ b1g,
                 const float* __restrict__ wW,
                 const float* __restrict__ node_feat,
                 const float* __restrict__ blk3,   // edge_feat (edge pass) / edge_out (node pass)
                 const float* __restrict__ rbf,
                 const int*   __restrict__ src,
                 const int*   __restrict__ dstv,
                 float*       __restrict__ out,
                 int E) {
  extern __shared__ char smem[];
  __bf16* L0f = (__bf16*)(smem + OFF_L0);
  __bf16* G0f = (__bf16*)(smem + OFF_G0);
  __bf16* L1f = (__bf16*)(smem + OFF_L1);
  __bf16* G1f = (__bf16*)(smem + OFF_G1);
  float*  wWl = (float*)(smem + OFF_WW);
  float*  bls = (float*)(smem + OFF_B);

  const int tid = threadIdx.x, nthr = blockDim.x;
  stage_frag(w0l, L0f, GD_IN, tid, nthr);
  stage_frag(w0g, G0f, GD_IN, tid, nthr);
  stage_frag(w1l, L1f, GDIM,  tid, nthr);
  stage_frag(w1g, G1f, GDIM,  tid, nthr);
  for (int i = tid; i < 9 * GDIM; i += nthr) wWl[i] = wW[i];
  for (int i = tid; i < GDIM; i += nthr) {
    bls[i]            = b0l[i];
    bls[GDIM + i]     = b1l[i];
    bls[2 * GDIM + i] = b0g[i];
    bls[3 * GDIM + i] = b1g[i];
  }
  __syncthreads();

  const int wave = tid >> 5, lane = tid & 31;
  const int half = lane >> 4, ln = lane & 15;
  char* wbase = smem + OFF_WV + wave * WV_SZ;
  __bf16* xlds = (__bf16*)wbase;
  float*  wlds = (float*)(wbase + X_SZ);
  __bf16* hlds = (__bf16*)(wbase + X_SZ + W_SZ);

  const long ntiles = ((long)E + 15) >> 4;
  for (long tile = (long)blockIdx.x * WAVES + wave; tile < ntiles;
       tile += (long)gridDim.x * WAVES) {
    const long e0 = tile << 4;

    // ---- assemble x tile [16 x 192] as bf16: [vi | vj | blk3] ----
    for (int c4 = lane; c4 < 16 * 48; c4 += 32) {
      int m = c4 / 48, q = c4 - (c4 / 48) * 48;   // q-th float4 in the row
      long e = e0 + m;
      float4 v = make_float4(0.f, 0.f, 0.f, 0.f);
      if (e < E) {
        const float* p;
        if (q < 16)      p = node_feat + (size_t)src[e]  * GDIM + q * 4;
        else if (q < 32) p = node_feat + (size_t)dstv[e] * GDIM + (q - 16) * 4;
        else             p = blk3 + (size_t)e * GDIM + (q - 32) * 4;
        v = *(const float4*)p;
      }
      __bf16* xr = xlds + m * GD_IN + q * 4;
      xr[0] = (__bf16)v.x; xr[1] = (__bf16)v.y;
      xr[2] = (__bf16)v.z; xr[3] = (__bf16)v.w;
    }

    // ---- rbf @ wW tile [16 x 64] in f32 (K=9, VALU) ----
    for (int c = lane; c < 16 * GDIM; c += 32) {
      int m = c >> 6, n = c & 63;
      long e = e0 + m;
      float s = 0.f;
      if (e < E) {
        const float* rb = rbf + (size_t)e * 9;
        #pragma unroll
        for (int q = 0; q < 9; ++q) s += rb[q] * wWl[q * GDIM + n];
      }
      wlds[(m << 6) + n] = s;
    }
    asm volatile("s_wait_dscnt 0" ::: "memory");

    v8f pl[4], pg[4];
    mlp_branch(xlds, hlds, L0f, bls,            L1f, bls + GDIM,     false, lane, pl);
    mlp_branch(xlds, hlds, G0f, bls + 2 * GDIM, G1f, bls + 3 * GDIM, true,  lane, pg);

    // ---- epilogue ----
    #pragma unroll
    for (int nt = 0; nt < 4; ++nt) {
      #pragma unroll
      for (int r = 0; r < 8; ++r) {
        int m = r + half * 8;
        int n = nt * 16 + ln;
        long e = e0 + m;
        if (e < E) {
          float val = pl[nt][r] * pg[nt][r] * wlds[(m << 6) + n];
          if (ATOMIC) {
            atomicAdd(out + (size_t)dstv[e] * GDIM + n, val);    // segment_sum at dst
          } else {
            out[(size_t)e * GDIM + n] = blk3[(size_t)e * GDIM + n] + val;  // residual
          }
        }
      }
    }
  }
}

__global__ void copy_nodes_kernel(const float4* __restrict__ in,
                                  float4* __restrict__ out, int n4) {
  int i = blockIdx.x * blockDim.x + threadIdx.x;
  if (i < n4) out[i] = in[i];
}

extern "C" void kernel_launch(void* const* d_in, const int* in_sizes, int n_in,
                              void* d_out, int out_size, void* d_ws, size_t ws_size,
                              hipStream_t stream) {
  (void)n_in; (void)out_size; (void)d_ws; (void)ws_size;
  const float* e_lw0 = (const float*)d_in[0];
  const float* e_lb0 = (const float*)d_in[1];
  const float* e_lw1 = (const float*)d_in[2];
  const float* e_lb1 = (const float*)d_in[3];
  const float* e_gw0 = (const float*)d_in[4];
  const float* e_gb0 = (const float*)d_in[5];
  const float* e_gw1 = (const float*)d_in[6];
  const float* e_gb1 = (const float*)d_in[7];
  const float* e_wW  = (const float*)d_in[8];
  const float* n_lw0 = (const float*)d_in[9];
  const float* n_lb0 = (const float*)d_in[10];
  const float* n_lw1 = (const float*)d_in[11];
  const float* n_lb1 = (const float*)d_in[12];
  const float* n_gw0 = (const float*)d_in[13];
  const float* n_gb0 = (const float*)d_in[14];
  const float* n_gw1 = (const float*)d_in[15];
  const float* n_gb1 = (const float*)d_in[16];
  const float* n_wW  = (const float*)d_in[17];
  const float* node_feat = (const float*)d_in[18];
  const float* edge_feat = (const float*)d_in[19];
  const float* rbf = (const float*)d_in[20];
  const int*   src = (const int*)d_in[21];
  const int*   dst = (const int*)d_in[22];

  const int N = in_sizes[18] / GDIM;
  const int E = in_sizes[21];
  float* edge_out = (float*)d_out;
  float* node_out = (float*)d_out + (size_t)E * GDIM;

  // node_out starts as node_feat (residual base for the scatter-add)
  int n4 = N * (GDIM / 4);
  copy_nodes_kernel<<<(n4 + 255) / 256, 256, 0, stream>>>(
      (const float4*)node_feat, (float4*)node_out, n4);

  long ntiles = ((long)E + 15) / 16;
  int blocks = (int)((ntiles + WAVES - 1) / WAVES);

  // Pass 1: edge update -> edge_out
  gmlp_kernel<false><<<blocks, WAVES * 32, LDS_TOTAL, stream>>>(
      e_lw0, e_lb0, e_lw1, e_lb1, e_gw0, e_gb0, e_gw1, e_gb1, e_wW,
      node_feat, edge_feat, rbf, src, dst, edge_out, E);

  // Pass 2: node messages on [vi|vj|edge_out] -> atomic scatter into node_out
  gmlp_kernel<true><<<blocks, WAVES * 32, LDS_TOTAL, stream>>>(
      n_lw0, n_lb0, n_lw1, n_lb1, n_gw0, n_gb0, n_gw1, n_gb1, n_wW,
      node_feat, edge_out, rbf, src, dst, node_out, E);
}
